// DecoderGRU_30064771072107
// MI455X (gfx1250) — matrix operand order repaired and checked by reading, not compiled, not run on hardware
//
#include <hip/hip_runtime.h>

// Problem constants (match reference)
constexpr int kB = 8, kT = 128, kU = 40, kW = 1280;
constexpr int kVOCAB = 32000, kEMB = 256, kDEC = 512, kMEM = 512;

typedef unsigned short u16;
typedef __attribute__((ext_vector_type(16))) __bf16 v16bf;
typedef __attribute__((ext_vector_type(8)))  float  v8f;

// ---------------- bf16 helpers (round-to-nearest-even) ----------------------
__device__ __forceinline__ unsigned pk_bf16(float lo, float hi) {
  unsigned a = __float_as_uint(lo); a = a + 0x7FFFu + ((a >> 16) & 1u);
  unsigned b = __float_as_uint(hi); b = b + 0x7FFFu + ((b >> 16) & 1u);
  return (a >> 16) | (b & 0xFFFF0000u);
}
__device__ __forceinline__ u16 bf16s(float x) {
  unsigned a = __float_as_uint(x); a = a + 0x7FFFu + ((a >> 16) & 1u);
  return (u16)(a >> 16);
}

struct Frag { union { unsigned u[8]; v16bf v; }; };

// A fragment (16-bit A 16x32 ISA layout): lane holds halfs K=k+base..+7 and
// K=k+16+base..+7, base = (lane>=16)*8.  Two 16B loads.
__device__ __forceinline__ void load_afrag(Frag& f, const u16* row, int k, int base) {
  uint4 a = *(const uint4*)(row + k + base);
  uint4 b = *(const uint4*)(row + k + 16 + base);
  f.u[0] = a.x; f.u[1] = a.y; f.u[2] = a.z; f.u[3] = a.w;
  f.u[4] = b.x; f.u[5] = b.y; f.u[6] = b.z; f.u[7] = b.w;
}
// B fragment (32x16): halfs K=k+hi*8..+7 and K=k+16+hi*8..+7 of column n. Two 16B loads.
__device__ __forceinline__ void load_bfrag(Frag& f, const u16* row, int k, int hi) {
  uint4 a = *(const uint4*)(row + k + hi * 8);
  uint4 b = *(const uint4*)(row + k + 16 + hi * 8);
  f.u[0] = a.x; f.u[1] = a.y; f.u[2] = a.z; f.u[3] = a.w;
  f.u[4] = b.x; f.u[5] = b.y; f.u[6] = b.z; f.u[7] = b.w;
}
__device__ __forceinline__ v8f wmma_bf16(const Frag& a, const Frag& b, v8f c) {
  return __builtin_amdgcn_wmma_f32_16x16x32_bf16(
      false, a.v, false, b.v, (short)0, c, false, false);
}

// ---------------- generic WMMA GEMM: C = A(MxK)bf16 * Bw(NxK)bf16^T (+bias) -
// blockDim = 128 (4 waves). Each wave computes a 32x32 tile (4 accumulators,
// 2 A frags x 2 B frags -> 4 wmma per K chunk). Software-pipelined 1 deep.
// OOB rows are clamped (branchless); stores guarded.
template <int K, bool OUT_BF16>
__global__ __launch_bounds__(128) void k_gemm(
    const u16* __restrict__ A, const u16* __restrict__ Bw,
    const float* __restrict__ bias, void* __restrict__ Cv,
    int M, int N, int lda, int ldb, int ldc,
    long long sA, long long sB, long long sC) {
  const int wave = threadIdx.x >> 5, lane = threadIdx.x & 31;
  const int r = lane & 15, hi = lane >> 4, base = hi * 8;
  A  += (long long)blockIdx.z * sA;
  Bw += (long long)blockIdx.z * sB;
  const int n0 = blockIdx.x * 128 + wave * 32;
  const int m0 = blockIdx.y * 32;
  const u16* ap0 = A  + (size_t)min(m0 + r,      M - 1) * lda;
  const u16* ap1 = A  + (size_t)min(m0 + 16 + r, M - 1) * lda;
  const u16* bp0 = Bw + (size_t)min(n0 + r,      N - 1) * ldb;
  const u16* bp1 = Bw + (size_t)min(n0 + 16 + r, N - 1) * ldb;

  v8f acc00 = {}, acc01 = {}, acc10 = {}, acc11 = {};
  Frag fa0, fa1, fb0, fb1;
  load_afrag(fa0, ap0, 0, base);
  load_afrag(fa1, ap1, 0, base);
  load_bfrag(fb0, bp0, 0, hi);
  load_bfrag(fb1, bp1, 0, hi);
#pragma unroll
  for (int k = 0; k < K; k += 32) {
    const int kn = (k + 32 < K) ? k + 32 : k;          // last iter: harmless reload
    Frag na0, na1, nb0, nb1;
    load_afrag(na0, ap0, kn, base);
    load_afrag(na1, ap1, kn, base);
    load_bfrag(nb0, bp0, kn, hi);
    load_bfrag(nb1, bp1, kn, hi);
    if (k + 256 < K) {
      __builtin_prefetch((const void*)(bp0 + k + 256), 0, 1);
      __builtin_prefetch((const void*)(bp1 + k + 256), 0, 1);
    }
    acc00 = wmma_bf16(fa0, fb0, acc00);
    acc01 = wmma_bf16(fa0, fb1, acc01);
    acc10 = wmma_bf16(fa1, fb0, acc10);
    acc11 = wmma_bf16(fa1, fb1, acc11);
    fa0 = na0; fa1 = na1; fb0 = nb0; fb1 = nb1;
  }

#pragma unroll
  for (int mh = 0; mh < 2; ++mh) {
#pragma unroll
    for (int nh = 0; nh < 2; ++nh) {
      const v8f& acc = mh ? (nh ? acc11 : acc10) : (nh ? acc01 : acc00);
      const int n = n0 + nh * 16 + r;
      if (n < N) {
        const float bv = bias ? bias[n] : 0.f;
#pragma unroll
        for (int v = 0; v < 8; ++v) {
          int m = m0 + mh * 16 + hi * 8 + v;
          if (m < M) {
            float val = acc[v] + bv;
            if (OUT_BF16)
              ((u16*)Cv)[(long long)blockIdx.z * sC + (size_t)m * ldc + n] = bf16s(val);
            else
              ((float*)Cv)[(long long)blockIdx.z * sC + (size_t)m * ldc + n] = val;
          }
        }
      }
    }
  }
}

// ---------------- elementwise f32 -> bf16 conversion ------------------------
__global__ void k_cvt(const float2* __restrict__ in, unsigned* __restrict__ out,
                      long long npairs) {
  long long i = blockIdx.x * (long long)blockDim.x + threadIdx.x;
  if (i < npairs) { float2 v = in[i]; out[i] = pk_bf16(v.x, v.y); }
}

// ---------------- embedding gather (f32 emb -> bf16 x) ----------------------
__global__ void k_embed(const int* __restrict__ target, const float* __restrict__ emb,
                        u16* __restrict__ xb) {
  int bt = blockIdx.x;                                 // B*T blocks, 128 thr
  int tok = target[bt];
  const float2* src = (const float2*)(emb + (size_t)tok * kEMB);
  unsigned* dst = (unsigned*)(xb + (size_t)bt * kEMB);
  float2 v = src[threadIdx.x];
  dst[threadIdx.x] = pk_bf16(v.x, v.y);
}

// ---------------- h0 = W_red * enc_state + b_red ----------------------------
__global__ void k_h0(const float* __restrict__ u_output, const int* __restrict__ u_len,
                     const float* __restrict__ W_red, const float* __restrict__ b_red,
                     float* __restrict__ h0, u16* __restrict__ h0b) {
  int idx = blockIdx.x * blockDim.x + threadIdx.x;
  if (idx >= kB * kDEC) return;
  int b = idx / kDEC, d = idx % kDEC;
  int ul = u_len[b];
  const int half = kMEM / 2;
  const float* fwd = u_output + ((size_t)b * kU + (ul - 1)) * kMEM;
  const float* bwd = u_output + ((size_t)b * kU + 0) * kMEM + half;
  const float* wr  = W_red + (size_t)d * kMEM;
  float s = b_red[d];
  for (int k = 0; k < half; ++k) s += wr[k] * fwd[k];
  for (int k = 0; k < half; ++k) s += wr[half + k] * bwd[k];
  h0[idx] = s;
  h0b[idx] = bf16s(s);
}

// ---------------- grid barrier (persistent GRU) -----------------------------
__device__ __forceinline__ void grid_barrier(unsigned* cnt, unsigned* gen, unsigned nb) {
  __syncthreads();
  if (threadIdx.x == 0) {
    __threadfence();
    unsigned g = __hip_atomic_load(gen, __ATOMIC_ACQUIRE, __HIP_MEMORY_SCOPE_AGENT);
    if (__hip_atomic_fetch_add(cnt, 1u, __ATOMIC_ACQ_REL, __HIP_MEMORY_SCOPE_AGENT) == nb - 1u) {
      __hip_atomic_store(cnt, 0u, __ATOMIC_RELEASE, __HIP_MEMORY_SCOPE_AGENT);
      __hip_atomic_fetch_add(gen, 1u, __ATOMIC_ACQ_REL, __HIP_MEMORY_SCOPE_AGENT);
    } else {
      while (__hip_atomic_load(gen, __ATOMIC_ACQUIRE, __HIP_MEMORY_SCOPE_AGENT) == g)
        __builtin_amdgcn_s_sleep(1);
    }
  }
  __syncthreads();
}

// ---------------- persistent GRU scan ---------------------------------------
// 12 blocks x 256 threads (8 waves) -> 96 waves = 1536/16 N tiles of gh.
// W_hh fragments are loop-invariant -> compiler keeps them VGPR-resident.
// A-fragment loads are software-pipelined one chunk ahead.
__global__ __launch_bounds__(256) void k_gru(
    const float* __restrict__ gx, const float* __restrict__ h0,
    const u16* __restrict__ h0b, const u16* __restrict__ Whhb,
    const float* __restrict__ b_hh,
    float* __restrict__ rnn, u16* __restrict__ rnnb,
    float* __restrict__ gh, u16* __restrict__ ccb,
    unsigned* __restrict__ bar) {
  const unsigned NB = 12;
  const int wave = threadIdx.x >> 5, lane = threadIdx.x & 31;
  const int g = blockIdx.x * 8 + wave;                 // 0..95
  const int n0 = g * 16;
  const int r = lane & 15, hi = lane >> 4, base = hi * 8;
  const int rb = min(r, kB - 1);                       // clamp A row to valid batch
  const u16* wrow = Whhb + (size_t)(n0 + r) * kDEC;
  const float bv = b_hh[n0 + r];

  for (int t = 0; t < kT; ++t) {
    const u16* hrow = (t == 0) ? h0b + (size_t)rb * kDEC
                               : rnnb + ((size_t)rb * kT + (t - 1)) * kDEC;
    v8f acc = {};
    Frag fa;
    load_afrag(fa, hrow, 0, base);
#pragma unroll
    for (int k = 0; k < kDEC; k += 32) {
      const int kn = (k + 32 < kDEC) ? k + 32 : k;
      Frag na, fb;
      load_afrag(na, hrow, kn, base);
      load_bfrag(fb, wrow, k, hi);                     // invariant across t
      acc = wmma_bf16(fa, fb, acc);
      fa = na;
    }
    if (hi == 0) {                                     // rows m = v = batch index
#pragma unroll
      for (int v = 0; v < kB; ++v)
        gh[(size_t)v * (3 * kDEC) + n0 + r] = acc[v] + bv;
    }
    grid_barrier(bar, bar + 1, NB);

    // gate update: B*DEC = 4096 elements over 3072 threads
    for (int i = blockIdx.x * 256 + threadIdx.x; i < kB * kDEC; i += NB * 256) {
      int b = i >> 9, d = i & (kDEC - 1);
      const float* gxt = gx + ((size_t)b * kT + t) * (3 * kDEC);
      float ghr = gh[(size_t)b * 3 * kDEC + d];
      float ghz = gh[(size_t)b * 3 * kDEC + kDEC + d];
      float ghn = gh[(size_t)b * 3 * kDEC + 2 * kDEC + d];
      float rr = 1.f / (1.f + __expf(-(gxt[d] + ghr)));
      float zz = 1.f / (1.f + __expf(-(gxt[kDEC + d] + ghz)));
      float nn = tanhf(gxt[2 * kDEC + d] + rr * ghn);
      float hp = (t == 0) ? h0[(size_t)b * kDEC + d]
                          : rnn[((size_t)b * kT + t - 1) * kDEC + d];
      float hn = (1.f - zz) * nn + zz * hp;
      size_t o = ((size_t)b * kT + t) * kDEC + d;
      rnn[o] = hn;
      rnnb[o] = bf16s(hn);
      ccb[((size_t)b * kT + t) * (kDEC + kMEM) + kMEM + d] = bf16s(hn);
    }
    grid_barrier(bar, bar + 1, NB);
  }
}

// ---------------- pu = softmax over U with length mask (in place, f32) ------
__global__ __launch_bounds__(64) void k_pu(float* __restrict__ su,
                                           const int* __restrict__ u_len) {
  __shared__ float red[64];
  int bt = blockIdx.x, b = bt / kT;
  int ul = u_len[b];
  float* row = su + (size_t)bt * kU;
  int u = threadIdx.x;
  float v = (u < kU && u < ul) ? row[u] : -INFINITY;
  red[u] = v; __syncthreads();
  for (int s = 32; s > 0; s >>= 1) { if (u < s) red[u] = fmaxf(red[u], red[u + s]); __syncthreads(); }
  float m = red[0]; __syncthreads();
  if (m == -INFINITY) m = 0.f;
  float e = (u < kU && u < ul) ? __expf(v - m) : 0.f;
  red[u] = e; __syncthreads();
  for (int s = 32; s > 0; s >>= 1) { if (u < s) red[u] += red[u + s]; __syncthreads(); }
  float denom = fmaxf(red[0], 1e-30f);
  if (u < kU) row[u] = e / denom;
}

// ---------------- segment starts via binary search --------------------------
__global__ void k_segb(const int* __restrict__ seg_id, int* __restrict__ segs) {
  int idx = blockIdx.x * blockDim.x + threadIdx.x;     // B*(U+1)
  if (idx >= kB * (kU + 1)) return;
  int b = idx / (kU + 1), u = idx % (kU + 1);
  const int* s = seg_id + (size_t)b * kW;
  int lo = 0, hiW = kW;
  while (lo < hiW) { int mid = (lo + hiW) >> 1; if (s[mid] < u) lo = mid + 1; else hiW = mid; }
  segs[idx] = lo;
}

// ---------------- per-segment softmax * pu -> scores_uw (bf16) --------------
__global__ __launch_bounds__(128) void k_attn(const float* __restrict__ sw,
                                              const float* __restrict__ pu,
                                              const int* __restrict__ segs,
                                              const int* __restrict__ w_len,
                                              u16* __restrict__ swb) {
  int blk = blockIdx.x;                                // B*U blocks
  int b = blk / kU, u = blk % kU;
  int s0 = segs[b * (kU + 1) + u], s1 = segs[b * (kU + 1) + u + 1];
  int wl = w_len[b];
  int t = threadIdx.x;                                 // T = 128
  const float* row = sw + ((size_t)b * kT + t) * kW;
  u16* rowb = swb + ((size_t)b * kT + t) * kW;
  float m = -INFINITY;
  for (int w = s0; w < s1; ++w) if (w < wl) m = fmaxf(m, row[w]);
  float msafe = (m == -INFINITY) ? 0.f : m;
  float s = 0.f;
  for (int w = s0; w < s1; ++w) if (w < wl) s += __expf(row[w] - msafe);
  float p = pu[((size_t)b * kT + t) * kU + u];
  float inv = p / fmaxf(s, 1e-30f);
  for (int w = s0; w < s1; ++w) {
    float v = (w < wl) ? __expf(row[w] - msafe) * inv : 0.f;
    rowb[w] = bf16s(v);
  }
}

// ---------------- transpose w_output (B,W,MEM)f32 -> (B,MEM,W)bf16 ----------
__global__ void k_transpose(const float* __restrict__ in, u16* __restrict__ out) {
  __shared__ float tile[32][33];
  int b = blockIdx.z;
  int w0 = blockIdx.x * 32, m0 = blockIdx.y * 32;
  int tx = threadIdx.x, ty = threadIdx.y;              // 32 x 8
  for (int i = ty; i < 32; i += 8)
    tile[i][tx] = in[((size_t)b * kW + w0 + i) * kMEM + m0 + tx];
  __syncthreads();
  for (int i = ty; i < 32; i += 8)
    out[((size_t)b * kMEM + m0 + i) * kW + w0 + tx] = bf16s(tile[tx][i]);
}

// ---------------- row log-softmax over VOCAB (in place on d_out) ------------
__global__ __launch_bounds__(256) void k_lsm(float* __restrict__ out) {
  __shared__ float red[256];
  float* p = out + (size_t)blockIdx.x * kVOCAB;
  int tid = threadIdx.x;
  float m = -INFINITY;
  for (int i = tid; i < kVOCAB; i += 256) m = fmaxf(m, p[i]);
  red[tid] = m; __syncthreads();
  for (int s = 128; s > 0; s >>= 1) { if (tid < s) red[tid] = fmaxf(red[tid], red[tid + s]); __syncthreads(); }
  m = red[0]; __syncthreads();
  float sum = 0.f;
  for (int i = tid; i < kVOCAB; i += 256) sum += __expf(p[i] - m);
  red[tid] = sum; __syncthreads();
  for (int s = 128; s > 0; s >>= 1) { if (tid < s) red[tid] += red[tid + s]; __syncthreads(); }
  float lse = m + logf(red[0]);
  for (int i = tid; i < kVOCAB; i += 256) p[i] -= lse;
}

// ---------------- host helpers ----------------------------------------------
static inline void cvt(hipStream_t st, const float* in, u16* out, long long n) {
  long long np = n / 2;
  k_cvt<<<(unsigned)((np + 255) / 256), 256, 0, st>>>((const float2*)in, (unsigned*)out, np);
}

template <bool OUT_BF16>
static inline void gemm(hipStream_t st, const u16* A, const u16* Bw,
                        const float* bias, void* C, int M, int N, int K,
                        int lda, int ldb, int ldc,
                        long long sA, long long sB, long long sC, int batch) {
  dim3 grid((N + 127) / 128, (M + 31) / 32, batch);
  switch (K) {
    case 256:
      k_gemm<256, OUT_BF16><<<grid, 128, 0, st>>>(A, Bw, bias, C, M, N, lda, ldb, ldc, sA, sB, sC); break;
    case 512:
      k_gemm<512, OUT_BF16><<<grid, 128, 0, st>>>(A, Bw, bias, C, M, N, lda, ldb, ldc, sA, sB, sC); break;
    case 1024:
      k_gemm<1024, OUT_BF16><<<grid, 128, 0, st>>>(A, Bw, bias, C, M, N, lda, ldb, ldc, sA, sB, sC); break;
    case 1280:
      k_gemm<1280, OUT_BF16><<<grid, 128, 0, st>>>(A, Bw, bias, C, M, N, lda, ldb, ldc, sA, sB, sC); break;
    default: break;  // unused
  }
}

extern "C" void kernel_launch(void* const* d_in, const int* in_sizes, int n_in,
                              void* d_out, int out_size, void* d_ws, size_t ws_size,
                              hipStream_t stream) {
  (void)in_sizes; (void)n_in; (void)out_size; (void)ws_size;
  const int*   target   = (const int*)d_in[0];
  const int*   seg_id   = (const int*)d_in[1];
  const int*   u_len    = (const int*)d_in[2];
  const int*   w_len    = (const int*)d_in[3];
  const float* u_output = (const float*)d_in[4];
  const float* w_output = (const float*)d_in[5];
  const float* emb      = (const float*)d_in[6];
  const float* W_red    = (const float*)d_in[7];
  const float* b_red    = (const float*)d_in[8];
  const float* W_ih     = (const float*)d_in[9];
  const float* W_hh     = (const float*)d_in[10];
  const float* b_ih     = (const float*)d_in[11];
  const float* b_hh     = (const float*)d_in[12];
  const float* W_au     = (const float*)d_in[13];
  const float* b_au     = (const float*)d_in[14];
  const float* W_aw     = (const float*)d_in[15];
  const float* b_aw     = (const float*)d_in[16];
  const float* W_out    = (const float*)d_in[17];
  const float* b_out    = (const float*)d_in[18];
  float* out = (float*)d_out;

  char* ws = (char*)d_ws;
  size_t off = 0;
  auto alloc = [&](size_t bytes) -> void* {
    void* p = ws + off; off = (off + bytes + 255) & ~(size_t)255; return p;
  };
  // f32 intermediates
  float* gx  = (float*)alloc((size_t)kB * kT * 3 * kDEC * 4);
  float* h0  = (float*)alloc((size_t)kB * kDEC * 4);
  float* gh  = (float*)alloc((size_t)kB * 3 * kDEC * 4);
  float* rnn = (float*)alloc((size_t)kB * kT * kDEC * 4);
  float* su  = (float*)alloc((size_t)kB * kT * kU * 4);
  float* sw  = (float*)alloc((size_t)kB * kT * kW * 4);
  int*   segs = (int*)alloc((size_t)kB * (kU + 1) * 4);
  // bf16 operands
  u16* xb    = (u16*)alloc((size_t)kB * kT * kEMB * 2);
  u16* h0b   = (u16*)alloc((size_t)kB * kDEC * 2);
  u16* rnnb  = (u16*)alloc((size_t)kB * kT * kDEC * 2);
  u16* Wihb  = (u16*)alloc((size_t)3 * kDEC * kEMB * 2);
  u16* Whhb  = (u16*)alloc((size_t)3 * kDEC * kDEC * 2);
  u16* Waub  = (u16*)alloc((size_t)kDEC * kMEM * 2);
  u16* Wawb  = (u16*)alloc((size_t)kDEC * kMEM * 2);
  u16* Woutb = (u16*)alloc((size_t)kVOCAB * (kDEC + kMEM) * 2);
  u16* uob   = (u16*)alloc((size_t)kB * kU * kMEM * 2);
  u16* wob   = (u16*)alloc((size_t)kB * kW * kMEM * 2);
  u16* aub   = (u16*)alloc((size_t)kB * kU * kDEC * 2);
  u16* awb   = (u16*)alloc((size_t)kB * kW * kDEC * 2);
  u16* swb   = (u16*)alloc((size_t)kB * kT * kW * 2);
  u16* woTb  = (u16*)alloc((size_t)kB * kMEM * kW * 2);
  u16* ccb   = (u16*)alloc((size_t)kB * kT * (kDEC + kMEM) * 2);
  unsigned* bar = (unsigned*)alloc(256);

  hipMemsetAsync(bar, 0, 8, stream);

  // 0) one-shot f32 -> bf16 conversions (cheap, HBM-bound)
  cvt(stream, W_ih,  Wihb,  (long long)3 * kDEC * kEMB);
  cvt(stream, W_hh,  Whhb,  (long long)3 * kDEC * kDEC);
  cvt(stream, W_au,  Waub,  (long long)kDEC * kMEM);
  cvt(stream, W_aw,  Wawb,  (long long)kDEC * kMEM);
  cvt(stream, W_out, Woutb, (long long)kVOCAB * (kDEC + kMEM));
  cvt(stream, u_output, uob, (long long)kB * kU * kMEM);
  cvt(stream, w_output, wob, (long long)kB * kW * kMEM);

  // 1) embedding + h0
  k_embed<<<kB * kT, kEMB / 2, 0, stream>>>(target, emb, xb);
  k_h0<<<(kB * kDEC + 255) / 256, 256, 0, stream>>>(u_output, u_len, W_red, b_red, h0, h0b);

  // 2) gx = x @ W_ih^T + b_ih   (1024 x 1536 x 256)
  gemm<false>(stream, xb, Wihb, b_ih, gx, kB * kT, 3 * kDEC, kEMB,
              kEMB, kEMB, 3 * kDEC, 0, 0, 0, 1);

  // 3) GRU scan (persistent kernel, grid barrier between steps)
  k_gru<<<12, 256, 0, stream>>>(gx, h0, h0b, Whhb, b_hh, rnn, rnnb, gh, ccb, bar);

  // 4) attention projections -> bf16
  gemm<true>(stream, uob, Waub, b_au, aub, kB * kU, kDEC, kMEM,
             kMEM, kMEM, kDEC, 0, 0, 0, 1);
  gemm<true>(stream, wob, Wawb, b_aw, awb, kB * kW, kDEC, kMEM,
             kMEM, kMEM, kDEC, 0, 0, 0, 1);

  // 5) batched scores (f32 out for softmax)
  gemm<false>(stream, rnnb, aub, nullptr, su, kT, kU, kDEC, kDEC, kDEC, kU,
              (long long)kT * kDEC, (long long)kU * kDEC, (long long)kT * kU, kB);
  gemm<false>(stream, rnnb, awb, nullptr, sw, kT, kW, kDEC, kDEC, kDEC, kW,
              (long long)kT * kDEC, (long long)kW * kDEC, (long long)kT * kW, kB);

  // 6) pu softmax, segment bounds, segment softmax -> scores_uw (bf16)
  k_pu<<<kB * kT, 64, 0, stream>>>(su, u_len);
  k_segb<<<(kB * (kU + 1) + 255) / 256, 256, 0, stream>>>(seg_id, segs);
  k_attn<<<kB * kU, kT, 0, stream>>>(sw, su, segs, w_len, swb);

  // 7) context = scores_uw @ w_output (via bf16 transpose, K contiguous)
  k_transpose<<<dim3(kW / 32, kMEM / 32, kB), dim3(32, 8), 0, stream>>>(w_output, woTb);
  gemm<true>(stream, swb, woTb, nullptr, ccb, kT, kMEM, kW, kW, kW, kDEC + kMEM,
             (long long)kT * kW, (long long)kMEM * kW, (long long)kT * (kDEC + kMEM), kB);

  // 8) output GEMM: dec = concat @ W_out^T + b_out  (1024 x 32000 x 1024)
  gemm<false>(stream, ccb, Woutb, b_out, out, kB * kT, kVOCAB, kDEC + kMEM,
              kDEC + kMEM, kDEC + kMEM, kVOCAB, 0, 0, 0, 1);

  // 9) log_softmax over vocab, in place on d_out
  k_lsm<<<kB * kT, 256, 0, stream>>>(out);
}
